// DecoderSyntaxTreeNetwork_82136954569226
// MI455X (gfx1250) — matrix operand order repaired and checked by reading, not compiled, not run on hardware
//
#include <hip/hip_runtime.h>
#include <hip/hip_bf16.h>

#define H 1024
#define NT_OUT 2048
#define T_OUT 8192
#define R_NT 32
#define R_T 8
#define DEPTH 10
#define NLEAF 1024
// LDS A-tile row stride in bf16 elements: 1032*2B = 516 dwords == 4 (mod 64)
// -> lane m (row m) starts at bank 4m; 32 lanes x b128 cover 64 banks twice
// (minimum possible), instead of the 16-way conflict a 1024 stride causes.
#define AS_STRIDE 1032

typedef __attribute__((ext_vector_type(16))) __bf16 v16bf;
typedef __attribute__((ext_vector_type(8)))  float  v8f;

__device__ __forceinline__ unsigned short f32_to_bf16(float f) {
    unsigned int u = __builtin_bit_cast(unsigned int, f);
    u = u + 0x7FFFu + ((u >> 16) & 1u);   // round-to-nearest-even
    return (unsigned short)(u >> 16);
}

// ---------------------------------------------------------------------------
// Weight conversion f32 -> bf16 (once per call; makes Wnt fit in 192MB L2)
// ---------------------------------------------------------------------------
__global__ void cvt_f32_bf16(const float* __restrict__ src,
                             unsigned short* __restrict__ dst, size_t nVec4) {
    size_t i = (size_t)blockIdx.x * blockDim.x + threadIdx.x;
    size_t stride = (size_t)gridDim.x * blockDim.x;
    const float4* s4 = (const float4*)src;
    for (; i < nVec4; i += stride) {
        float4 f = s4[i];
        ushort4 o;
        o.x = f32_to_bf16(f.x); o.y = f32_to_bf16(f.y);
        o.z = f32_to_bf16(f.z); o.w = f32_to_bf16(f.w);
        ((ushort4*)dst)[i] = o;
    }
}

// ---------------------------------------------------------------------------
// Bucket nodes by rule, emit 16-row tile descriptors (pad slots = -1).
// Single block; n <= 1024, nRules <= 32, maxTiles <= 128.
// ---------------------------------------------------------------------------
__global__ void build_tiles(const int* __restrict__ rules, int n, int nRules,
                            int maxTiles, int* __restrict__ tileRule,
                            int* __restrict__ tileNodes) {
    __shared__ int cnt[R_NT];
    __shared__ int start[R_NT];
    __shared__ int tstart[R_NT];
    __shared__ int sorted[NLEAF];
    __shared__ int total;
    int tid = threadIdx.x;
    if (tid < nRules) cnt[tid] = 0;
    __syncthreads();
    int r = -1, p = 0;
    if (tid < n) { r = rules[tid]; p = atomicAdd(&cnt[r], 1); }
    __syncthreads();
    if (tid == 0) {
        int s = 0, t = 0;
        for (int i = 0; i < nRules; ++i) {
            start[i] = s; tstart[i] = t;
            s += cnt[i]; t += (cnt[i] + 15) >> 4;
        }
        total = t;
    }
    __syncthreads();
    if (tid < n) sorted[start[r] + p] = tid;
    __syncthreads();
    for (int t = tid; t < maxTiles; t += blockDim.x) {
        if (t >= total) {
            tileRule[t] = -1;
            for (int s = 0; s < 16; ++s) tileNodes[t * 16 + s] = -1;
            continue;
        }
        int tr = 0;
        for (int i = nRules - 1; i >= 0; --i)
            if (t >= tstart[i]) { tr = i; break; }
        int base = (t - tstart[tr]) << 4;
        tileRule[t] = tr;
        for (int s = 0; s < 16; ++s) {
            int idx = base + s;
            tileNodes[t * 16 + s] = (idx < cnt[tr]) ? sorted[start[tr] + idx] : -1;
        }
    }
}

// ---------------------------------------------------------------------------
// Shared GEMM core: one block = one 16-node tile x 64 output columns.
// 4 waves; each wave owns one 16x16 D tile, K=1024 in 32 bf16 WMMA steps.
// ---------------------------------------------------------------------------
__device__ __forceinline__ v8f tile_gemm_core(const float* __restrict__ hCur,
                                              const unsigned short* __restrict__ wrow,
                                              const int* __restrict__ tnodes,
                                              unsigned short* __restrict__ As,
                                              int lane) {
    int tid = threadIdx.x;
    // Stage A (16 rows x 1024 K) as bf16 into LDS, float4->ushort4 vectorized.
    for (int i = tid; i < 16 * (H / 4); i += 128) {
        int m = i >> 8;               // H/4 == 256 vec4 per row
        int kv = i & 255;
        int node = tnodes[m];
        float4 f = {0.f, 0.f, 0.f, 0.f};
        if (node >= 0) f = *(const float4*)(hCur + (size_t)node * H + kv * 4);
        ushort4 o;
        o.x = f32_to_bf16(f.x); o.y = f32_to_bf16(f.y);
        o.z = f32_to_bf16(f.z); o.w = f32_to_bf16(f.w);
        *(ushort4*)(As + m * AS_STRIDE + kv * 4) = o;
    }
    __syncthreads();

    int arow = lane & 15;
    int kb = (lane >> 4) << 4;             // 0 or 16 (contiguous K split, matches B)
    v8f acc = {0.f, 0.f, 0.f, 0.f, 0.f, 0.f, 0.f, 0.f};
    for (int kc = 0; kc < H; kc += 32) {
        __builtin_prefetch(wrow + kc + 128, 0, 1);   // global_prefetch_b8
        union { v16bf v; unsigned short s[16]; } A;
#pragma unroll
        for (int pp = 0; pp < 8; ++pp) {
            int k0 = kc + kb + (pp << 1);
            A.s[2 * pp]     = As[arow * AS_STRIDE + k0];
            A.s[2 * pp + 1] = As[arow * AS_STRIDE + k0 + 1];
        }
        v16bf B = *(const v16bf*)(const void*)(wrow + kc);   // 16 contiguous K
        acc = __builtin_amdgcn_wmma_f32_16x16x32_bf16(
            false, A.v, false, B, (short)0, acc, false, false);
    }
    return acc;
}

// Tree level: hNext[2*node+sel] = tanh(W[r] @ h[node] + Bnt[cn[node][sel]])
__global__ void __launch_bounds__(128)
tree_gemm(const float* __restrict__ hCur,
          const unsigned short* __restrict__ WntB,
          const float* __restrict__ Bnt,
          const int* __restrict__ cnLevel,
          const int* __restrict__ tileRule,
          const int* __restrict__ tileNodes,
          float* __restrict__ hNext) {
    __shared__ unsigned short As[16 * AS_STRIDE];
    int tile = blockIdx.x;
    int r = tileRule[tile];
    if (r < 0) return;
    const int* tnodes = tileNodes + tile * 16;
    int lane = threadIdx.x & 31, wave = threadIdx.x >> 5;
    int col = blockIdx.y * 64 + wave * 16 + (lane & 15);          // 0..2047
    const unsigned short* wrow =
        WntB + (size_t)r * NT_OUT * H + (size_t)col * H + ((lane >> 4) << 4);
    v8f acc = tile_gemm_core(hCur, wrow, tnodes, As, lane);

    int sel = (col >= H) ? 1 : 0;
    int kk = col & (H - 1);
#pragma unroll
    for (int v = 0; v < 8; ++v) {
        int m = v + ((lane >> 4) << 3);
        int node = tnodes[m];
        if (node >= 0) {
            float b = Bnt[cnLevel[node * 2 + sel] * H + kk];
            hNext[(size_t)(2 * node + sel) * H + kk] = tanhf(acc[v] + b);
        }
    }
}

// Head: out[leaf, col] = Wt[r] @ h[leaf]
__global__ void __launch_bounds__(128)
head_gemm(const float* __restrict__ hCur,
          const unsigned short* __restrict__ WtB,
          const int* __restrict__ tileRule,
          const int* __restrict__ tileNodes,
          float* __restrict__ out) {
    __shared__ unsigned short As[16 * AS_STRIDE];
    int tile = blockIdx.x;
    int r = tileRule[tile];
    if (r < 0) return;
    const int* tnodes = tileNodes + tile * 16;
    int lane = threadIdx.x & 31, wave = threadIdx.x >> 5;
    int col = blockIdx.y * 64 + wave * 16 + (lane & 15);          // 0..8191
    const unsigned short* wrow =
        WtB + (size_t)r * T_OUT * H + (size_t)col * H + ((lane >> 4) << 4);
    v8f acc = tile_gemm_core(hCur, wrow, tnodes, As, lane);
#pragma unroll
    for (int v = 0; v < 8; ++v) {
        int m = v + ((lane >> 4) << 3);
        int node = tnodes[m];
        if (node >= 0) out[(size_t)node * T_OUT + col] = acc[v];
    }
}

extern "C" void kernel_launch(void* const* d_in, const int* in_sizes, int n_in,
                              void* d_out, int out_size, void* d_ws, size_t ws_size,
                              hipStream_t stream) {
    const float* encoding    = (const float*)d_in[0];
    const int*   rules       = (const int*)d_in[1];
    const int*   child_nodes = (const int*)d_in[2];
    const int*   term_rules  = (const int*)d_in[3];
    const float* Wnt         = (const float*)d_in[4];
    const float* Wt          = (const float*)d_in[5];
    const float* Bnt         = (const float*)d_in[6];
    float* out = (float*)d_out;

    // workspace layout
    unsigned short* WntB = (unsigned short*)d_ws;                       // 128 MB
    unsigned short* WtB  = WntB + (size_t)R_NT * NT_OUT * H;            // 128 MB
    float* hA = (float*)(WtB + (size_t)R_T * T_OUT * H);                // 4 MB
    float* hB = hA + (size_t)NLEAF * H;                                 // 4 MB
    int* tileRule  = (int*)(hB + (size_t)NLEAF * H);
    int* tileNodes = tileRule + 128;

    cvt_f32_bf16<<<4096, 256, 0, stream>>>(Wnt, WntB, (size_t)R_NT * NT_OUT * H / 4);
    cvt_f32_bf16<<<4096, 256, 0, stream>>>(Wt,  WtB,  (size_t)R_T * T_OUT * H / 4);

    const float* hCur = encoding;           // level 0: one row
    float* bufs[2] = {hA, hB};
    int off = 0;
    for (int level = 0; level < DEPTH; ++level) {
        int n = 1 << level;
        int maxTiles = (n < R_NT ? n : R_NT) + (n >> 4);
        build_tiles<<<1, 1024, 0, stream>>>(rules + off, n, R_NT, maxTiles,
                                            tileRule, tileNodes);
        float* hNext = bufs[level & 1];
        dim3 grid(maxTiles, NT_OUT / 64);
        tree_gemm<<<grid, 128, 0, stream>>>(hCur, WntB, Bnt,
                                            child_nodes + 2 * off,
                                            tileRule, tileNodes, hNext);
        hCur = hNext;
        off += n;
    }
    int maxTilesF = R_T + (NLEAF >> 4);     // 72
    build_tiles<<<1, 1024, 0, stream>>>(term_rules, NLEAF, R_T, maxTilesF,
                                        tileRule, tileNodes);
    dim3 gridF(maxTilesF, T_OUT / 64);
    head_gemm<<<gridF, 128, 0, stream>>>(hCur, WtB, tileRule, tileNodes, out);
}
